// HawkesKT_5677946765617
// MI455X (gfx1250) — matrix operand order repaired and checked by reading, not compiled, not run on hardware
//
#include <hip/hip_runtime.h>

typedef __attribute__((ext_vector_type(16))) _Float16 v16h;
typedef __attribute__((ext_vector_type(8)))  _Float16 v8h;
typedef __attribute__((ext_vector_type(4)))  _Float16 v4h;
typedef __attribute__((ext_vector_type(8)))  float    v8f;
typedef __attribute__((ext_vector_type(4)))  float    v4f;
typedef __attribute__((ext_vector_type(4)))  int      v4i;

#define N_SKILLS_C 1000
#define EMB       256
#define SEQ       2048
// exp(-beta * ln(d)/ln5) == exp2(-beta * log2(d) * (log2 e / log2 5))
#define EXP2_SCALE 0.6213349345596119f
#define LOG2E      1.4426950408889634f

// f16 workspace table offsets (in halfs)
#define OFF_AI 0                      // alpha_inter: 2000*256
#define OFF_BI (2 * N_SKILLS_C * EMB) // beta_inter:  2000*256
#define OFF_AS (4 * N_SKILLS_C * EMB) // alpha_skill: 1000*256
#define OFF_BS (5 * N_SKILLS_C * EMB) // beta_skill:  1000*256

// ---------- one-shot f32 -> f16 table conversion ----------
__global__ __launch_bounds__(256)
void cvt_table_f16(const float* __restrict__ in, _Float16* __restrict__ out, int n4) {
    const int i = blockIdx.x * 256 + threadIdx.x;
    if (i < n4) {
        const v4f v = *(const v4f*)(in + i * 4);
        v4h h;
#pragma unroll
        for (int e = 0; e < 4; ++e) h[e] = (_Float16)v[e];
        *(v4h*)(out + i * 4) = h;
    }
}

// Assemble a 16x32 fragment: per-lane 16 halfs = piece0 (K k0..k0+7) ++ piece1 (K k0+16..k0+23)
__device__ __forceinline__ v16h load_frag(const _Float16* __restrict__ row, int k0) {
    const v8h p0 = *(const v8h*)(row + k0);
    const v8h p1 = *(const v8h*)(row + k0 + 16);
    return __builtin_shufflevector(p0, p1, 0, 1, 2, 3, 4, 5, 6, 7,
                                           8, 9, 10, 11, 12, 13, 14, 15);
}

// ---------- fused HawkesKT kernel ----------
__global__ __launch_bounds__(128)
void hawkeskt_fused_wmma(const int*      __restrict__ skills,
                         const int*      __restrict__ problems,
                         const int*      __restrict__ times,
                         const int*      __restrict__ labels,
                         const _Float16* __restrict__ aInter,   // f16 tables (pre-converted)
                         const _Float16* __restrict__ bInter,
                         const _Float16* __restrict__ aSkill,
                         const _Float16* __restrict__ bSkill,
                         const float*    __restrict__ problem_base_W,
                         const float*    __restrict__ skill_base_W,
                         float*          __restrict__ out)
{
    const int tTile = blockIdx.x & 127;   // S/16 = 128 target tiles
    const int b     = blockIdx.x >> 7;
    const int tBase = tTile << 4;

    const int tid  = threadIdx.x;
    const int wave = tid >> 5;
    const int lane = tid & 31;
    const int n    = lane & 15;           // column inside 16x16 tile
    const int hi   = lane >> 4;           // half-wave select

    __shared__ _Float16 ldsA[16 * EMB] __attribute__((aligned(32)));  // alpha tgt rows
    __shared__ _Float16 ldsB[16 * EMB] __attribute__((aligned(32)));  // beta  tgt rows
    __shared__ float    red[4 * 32];

    // ---- cooperative gather of the 16 target rows (already f16) ----
    // 16 rows x 256 halfs = 512 groups of 8 halfs across 128 threads (4 iters)
    for (int i = tid; i < 512; i += 128) {
        const int row = i >> 5;            // 32 groups per row
        const int c8  = (i & 31) << 3;
        const int sk  = skills[b * SEQ + tBase + row];
        *(v8h*)(&ldsA[row * EMB + c8]) = *(const v8h*)(aSkill + sk * EMB + c8);
        *(v8h*)(&ldsB[row * EMB + c8]) = *(const v8h*)(bSkill + sk * EMB + c8);
    }
    __syncthreads();

    // target time for this lane's column (fixed for the whole block)
    const int   t_glob = tBase + n;
    const float t_tgt  = (float)times[b * SEQ + t_glob] * 0.001f;

    float out_acc = 0.0f;

    // ---- walk source tiles of the strict lower triangle (s < t) ----
    for (int sTile = wave; sTile <= tTile; sTile += 4) {
        const int sBase = sTile << 4;
        const int srow  = sBase + n;                       // A row M = lane%16
        const int sk    = skills[b * SEQ + srow];
        const int it    = sk + labels[b * SEQ + srow] * N_SKILLS_C;
        const _Float16* arow = aInter + it * EMB;
        const _Float16* brow = bInter + it * EMB;

        v8f acc_a = {};
        v8f acc_b = {};

#pragma unroll
        for (int kc = 0; kc < 8; ++kc) {
            const int k0 = kc * 32 + hi * 8;
            // A fragments: two b128 loads each, no conversion
            const v16h Aa = load_frag(arow, k0);
            const v16h Ab = load_frag(brow, k0);
            // B fragments from LDS: column n = lane%16, K k0..k0+7 ++ k0+16..k0+23
            const int boff = n * EMB;
            const v16h Ba = load_frag(&ldsA[boff], k0);
            const v16h Bb = load_frag(&ldsB[boff], k0);

            acc_a = __builtin_amdgcn_wmma_f32_16x16x32_f16(
                        false, Aa, false, Ba, (short)0, acc_a, false, false);
            acc_b = __builtin_amdgcn_wmma_f32_16x16x32_f16(
                        false, Ab, false, Bb, (short)0, acc_b, false, false);
        }

        // ---- fused pointwise epilogue on the 16x16 accumulator tile ----
        // D layout: VGPR r, this lane -> M = r + 8*hi (source), N = n (target)
        const int  sm0 = sBase + hi * 8;
        const v4i  ts0 = *(const v4i*)(times + b * SEQ + sm0);
        const v4i  ts1 = *(const v4i*)(times + b * SEQ + sm0 + 4);

#pragma unroll
        for (int r = 0; r < 8; ++r) {
            const int s_glob = sm0 + r;
            if (s_glob < t_glob) {
                const int   tv    = (r < 4) ? ts0[r] : ts1[r & 3];
                const float t_src = (float)tv * 0.001f;
                const float alpha = acc_a[r];
                float beta = acc_b[r] + 1.0f;
                beta = fminf(fmaxf(beta, 0.0f), 10.0f);
                const float dt = fabsf(t_src - t_tgt);
                const float l2 = __builtin_amdgcn_logf(dt + 1e-10f);   // v_log_f32 (log2)
                out_acc += alpha * __builtin_amdgcn_exp2f(-beta * l2 * EXP2_SCALE); // v_exp_f32
            }
        }
    }

    // ---- reduce 8 (wave, half) partials per target, epilogue, store ----
    red[wave * 32 + lane] = out_acc;
    __syncthreads();
    if (tid < 16) {
        float s = 0.0f;
#pragma unroll
        for (int w = 0; w < 4; ++w)
            s += red[w * 32 + tid] + red[w * 32 + 16 + tid];
        const int gi = b * SEQ + tBase + tid;
        const float h = problem_base_W[problems[gi]] + skill_base_W[skills[gi]] + s;
        // sigmoid(h) = 1 / (1 + exp2(-h * log2 e))
        out[gi] = 1.0f / (1.0f + __builtin_amdgcn_exp2f(-h * LOG2E));
    }
}

extern "C" void kernel_launch(void* const* d_in, const int* in_sizes, int n_in,
                              void* d_out, int out_size, void* d_ws, size_t ws_size,
                              hipStream_t stream) {
    (void)in_sizes; (void)n_in; (void)ws_size; (void)out_size;
    const int*   skills        = (const int*)  d_in[0];
    const int*   problems      = (const int*)  d_in[1];
    const int*   times         = (const int*)  d_in[2];
    const int*   labels        = (const int*)  d_in[3];
    const float* alpha_inter_W = (const float*)d_in[4];
    const float* alpha_skill_W = (const float*)d_in[5];
    const float* beta_inter_W  = (const float*)d_in[6];
    const float* beta_skill_W  = (const float*)d_in[7];
    const float* problem_base_W= (const float*)d_in[8];
    const float* skill_base_W  = (const float*)d_in[9];
    float* out = (float*)d_out;

    _Float16* ws16 = (_Float16*)d_ws;      // 1,536,000 halfs = 3 MB
    _Float16* AI = ws16 + OFF_AI;
    _Float16* BI = ws16 + OFF_BI;
    _Float16* AS = ws16 + OFF_AS;
    _Float16* BS = ws16 + OFF_BS;

    // ---- stage 1: convert the four embedding tables to f16 (L2-resident) ----
    const int nInter4 = (2 * N_SKILLS_C * EMB) / 4;   // 128000
    const int nSkill4 = (N_SKILLS_C * EMB) / 4;       // 64000
    hipLaunchKernelGGL(cvt_table_f16, dim3((nInter4 + 255) / 256), dim3(256), 0, stream,
                       alpha_inter_W, AI, nInter4);
    hipLaunchKernelGGL(cvt_table_f16, dim3((nInter4 + 255) / 256), dim3(256), 0, stream,
                       beta_inter_W, BI, nInter4);
    hipLaunchKernelGGL(cvt_table_f16, dim3((nSkill4 + 255) / 256), dim3(256), 0, stream,
                       alpha_skill_W, AS, nSkill4);
    hipLaunchKernelGGL(cvt_table_f16, dim3((nSkill4 + 255) / 256), dim3(256), 0, stream,
                       beta_skill_W, BS, nSkill4);

    // ---- stage 2: fused flash-style WMMA kernel ----
    const int B = 32, Stiles = SEQ / 16;              // 32 * 128 = 4096 blocks
    hipLaunchKernelGGL(hawkeskt_fused_wmma, dim3(B * Stiles), dim3(128), 0, stream,
                       skills, problems, times, labels,
                       AI, BI, AS, BS,
                       problem_base_W, skill_base_W, out);
}
